// Net_20452634263740
// MI455X (gfx1250) — matrix operand order repaired and checked by reading, not compiled, not run on hardware
//
#include <hip/hip_runtime.h>

typedef __attribute__((ext_vector_type(16))) _Float16 v16h;
typedef __attribute__((ext_vector_type(8)))  _Float16 v8h;
typedef __attribute__((ext_vector_type(8)))  float    v8f;
typedef __attribute__((ext_vector_type(4)))  unsigned int u32x4;
typedef __attribute__((ext_vector_type(8)))  int      i32x8;
typedef __attribute__((ext_vector_type(4)))  int      i32x4;

#define B_   4
#define NANG 25
#define HW   1024

#if defined(__HIP_DEVICE_COMPILE__) && __has_builtin(__builtin_amdgcn_tensor_load_to_lds) && __has_builtin(__builtin_amdgcn_s_wait_tensorcnt)
#define HAVE_TDM 1
#else
#define HAVE_TDM 0
#endif

// ---------------------------------------------------------------------------
// Stage a contiguous f16 weight block into LDS. Preferred path: one TDM
// descriptor issued by wave 0 (Tensor Data Mover, tracked by TENSORcnt),
// fallback: cooperative vector copy.
// ---------------------------------------------------------------------------
__device__ __forceinline__ void stage_b_lds(_Float16* sdst, const _Float16* __restrict__ gsrc,
                                            unsigned nelem /* f16 elements, mult of 8, <= 16384 */)
{
#if HAVE_TDM
    if (threadIdx.x < 32) {
        unsigned long long ga = (unsigned long long)(uintptr_t)gsrc;
        unsigned lds_off = (unsigned)(uintptr_t)sdst;      // low 32 bits = DS byte offset
        u32x4 g0 = {};
        g0.x = 1u;                                          // count=1 (valid user descriptor)
        g0.y = lds_off;                                     // lds_addr
        g0.z = (unsigned)ga;                                // global_addr[31:0]
        g0.w = (unsigned)((ga >> 32) & 0x01ffffffu) | (2u << 30);  // global_addr[56:32] | type=2
        i32x8 g1 = {};
        g1[0] = 1 << 16;                                    // data_size=1 -> 2-byte elements
        g1[1] = (int)((nelem & 0xffffu) << 16);             // tensor_dim0[15:0]   (bits 63:48)
        g1[2] = (int)(((nelem >> 16) & 0xffffu) | (1u << 16)); // tensor_dim0[31:16] | tensor_dim1=1
        g1[3] = (int)((nelem & 0xffffu) << 16);             // tile_dim0           (bits 127:112)
        g1[5] = (int)nelem;                                 // tensor_dim0_stride[31:0]
        i32x4 zz = {};
        i32x8 z8 = {};
        __builtin_amdgcn_tensor_load_to_lds(g0, g1, zz, zz, z8, 0);
        __builtin_amdgcn_s_wait_tensorcnt(0);
    }
    __syncthreads();
#else
    for (unsigned i = threadIdx.x; i < nelem / 8; i += blockDim.x)
        ((v8h*)sdst)[i] = ((const v8h*)gsrc)[i];
    __syncthreads();
#endif
}

// ---------------------------------------------------------------------------
// NT adjacent 16x16 WMMA tiles sharing one A fragment:
// C[t] = A[m0:m0+16, :] * B^T[n0+16t : n0+16t+16, :], K = 32*ksteps.
// Operand layouts per CDNA5 ISA for V_WMMA_F32_16X16X32_F16.
// ---------------------------------------------------------------------------
template <int NT>
__device__ __forceinline__ void wmma_tiles(const _Float16* __restrict__ A, int lda, int m0,
                                           const _Float16* B, int ldb, int n0,
                                           int ksteps, v8f* acc)
{
    const int lane = threadIdx.x & 31;
    const int lo = lane & 15, hi = lane >> 4;
    const _Float16* Ar = A + (size_t)(m0 + lo) * lda;
    const _Float16* Br0 = B + (size_t)(n0 + lo) * ldb;
    for (int ks = 0; ks < ksteps; ++ks) {
        const int ka = ks * 32 + hi * 8;
        v8h a0 = *(const v8h*)(Ar + ka);
        v8h a1 = *(const v8h*)(Ar + ka + 16);
        v16h a;
#pragma unroll
        for (int i = 0; i < 8; ++i) { a[i] = a0[i]; a[i + 8] = a1[i]; }
        const int kb = ks * 32 + hi * 8;
#pragma unroll
        for (int t = 0; t < NT; ++t) {
            const _Float16* Br = Br0 + (size_t)(t * 16) * ldb;
            v8h b0 = *(const v8h*)(Br + kb);
            v8h b1 = *(const v8h*)(Br + kb + 16);
            v16h b;
#pragma unroll
            for (int i = 0; i < 8; ++i) { b[i] = b0[i]; b[i + 8] = b1[i]; }
            acc[t] = __builtin_amdgcn_wmma_f32_16x16x32_f16(false, a, false, b, (short)0, acc[t],
                                                            false, false);
        }
    }
}

// Flat index into a (b, 128, 25, 32, 32) tensor for folded coordinate (rm, col).
__device__ __forceinline__ int fold_src_index(int fold, int b, int rm, int col)
{
    int ch, n, hh, ww;
    if (fold == 0) {                 // sf: M = cc(32)*h(32), K = w*25 + n
        ch = rm >> 5; hh = rm & 31;
        n = col % 25; ww = col / 25;
    } else if (fold == 1) {          // af: M = cc(32)*u(5), K = h*160 + v*32 + w
        ch = 32 + rm / 5; int u = rm % 5;
        hh = col / 160; int v5 = (col >> 5) % 5; ww = col & 31;
        n = u * 5 + v5;
    } else {                         // sa: M = h(32)*u(5), K = cc(64)*160 + v*32 + w
        hh = rm / 5; int u = rm % 5;
        ch = 64 + col / 160; int v5 = (col >> 5) % 5; ww = col & 31;
        n = u * 5 + v5;
    }
    return ((b * 128 + ch) * 25 + n) * 1024 + hh * 32 + ww;
}

// ---------------------------------------------------------------------------
// Prep kernels
// ---------------------------------------------------------------------------
__global__ void k_transpose_f16(const float* __restrict__ src, _Float16* __restrict__ dst,
                                int rows, int cols)
{
    int i = blockIdx.x * 256 + threadIdx.x;
    if (i >= rows * cols) return;
    int k = i / cols, n = i % cols;
    dst[(size_t)n * rows + k] = (_Float16)src[i];
}

__global__ void k_tok(const float* __restrict__ x, _Float16* __restrict__ xt)
{
    int i = blockIdx.x * 256 + threadIdx.x;     // over 4*64*25*1024 = 6553600
    int m = i & 1023;
    int n = (i >> 10) % 25;
    int k = (i / (25 * 1024)) % 64;
    int b = i / (64 * 25 * 1024);
    int t = (b * 25 + n) * 1024 + m;
    xt[(size_t)t * 64 + k] = (_Float16)x[i];
}

// ---------------------------------------------------------------------------
// Projection: out(b,128,n,hw) = xtok(T x 64) @ W^T; W^T staged in LDS via TDM.
// Block: 8 waves = 4 mtiles x 2 ngroups(4 tiles each). Grid 1600.
// ---------------------------------------------------------------------------
__global__ void k_proj(const _Float16* __restrict__ xtok, const _Float16* __restrict__ wt,
                       float* __restrict__ out)
{
    __shared__ __align__(16) _Float16 swb[128 * 64];
    stage_b_lds(swb, wt, 128 * 64);
    int wv = threadIdx.x >> 5;
    int mt = blockIdx.x * 4 + (wv & 3);
    int ng = wv >> 2;
    v8f acc[4] = {};
    wmma_tiles<4>(xtok, 64, mt * 16, swb, 64, ng * 64, 2, acc);
    int lane = threadIdx.x & 31, lo = lane & 15, hi = lane >> 4;
#pragma unroll
    for (int t = 0; t < 4; ++t) {
        int ch = ng * 64 + t * 16 + lo;
#pragma unroll
        for (int r = 0; r < 8; ++r) {
            int m = mt * 16 + r + hi * 8;
            int b = m / 25600, n = (m >> 10) % 25, hw = m & 1023;
            out[(size_t)((b * 128 + ch) * 25 + n) * 1024 + hw] = acc[t][r];
        }
    }
}

// ---------------------------------------------------------------------------
// Fold gather: (b,128,25,32,32) f32 -> folded f16 (optional l2-norm / transpose)
// ---------------------------------------------------------------------------
__global__ void k_fold(const float* __restrict__ src, _Float16* __restrict__ dst,
                       int fold, int Mf, int Kf, int normalize, int transpose)
{
    int row = blockIdx.x;
    int b = row / Mf, rm = row % Mf;
    int tid = threadIdx.x;
    __shared__ float red[256];
    float scale = 1.0f;
    if (normalize) {
        float ss = 0.f;
        for (int col = tid; col < Kf; col += 256) {
            float v = src[fold_src_index(fold, b, rm, col)];
            ss += v * v;
        }
        red[tid] = ss; __syncthreads();
        for (int s = 128; s > 0; s >>= 1) { if (tid < s) red[tid] += red[tid + s]; __syncthreads(); }
        scale = 1.f / fmaxf(sqrtf(red[0]), 1e-12f);
        __syncthreads();
    }
    _Float16* db = dst + (size_t)b * Mf * Kf;
    for (int col = tid; col < Kf; col += 256) {
        float v = src[fold_src_index(fold, b, rm, col)] * scale;
        size_t di = transpose ? ((size_t)col * Mf + rm) : ((size_t)rm * Kf + col);
        db[di] = (_Float16)v;
    }
}

// ---------------------------------------------------------------------------
// Scores: S = Qf @ Kf^T, causal tile-skip, 2 n-tiles per wave.
// ---------------------------------------------------------------------------
__global__ void k_scores(const _Float16* __restrict__ qf, const _Float16* __restrict__ kf,
                         float* __restrict__ S, int Mf, int Kf)
{
    int wave = blockIdx.x * 8 + (threadIdx.x >> 5);
    int nmt = Mf >> 4, ngc = nmt >> 1;
    int per_b = nmt * ngc;
    int b = wave / per_b, r = wave % per_b;
    int mt = r / ngc, nt0 = (r % ngc) * 2;
    if (nt0 > mt) return;                          // both tiles fully masked
    const _Float16* qb = qf + (size_t)b * Mf * Kf;
    const _Float16* kb = kf + (size_t)b * Mf * Kf;
    v8f acc[2] = {};
    wmma_tiles<2>(qb, Kf, mt * 16, kb, Kf, nt0 * 16, Kf >> 5, acc);
    float* Sb = S + (size_t)b * Mf * Mf;
    int lane = threadIdx.x & 31, lo = lane & 15, hi = lane >> 4;
#pragma unroll
    for (int t = 0; t < 2; ++t)
#pragma unroll
        for (int r2 = 0; r2 < 8; ++r2) {
            int m = mt * 16 + r2 + hi * 8, ncol = (nt0 + t) * 16 + lo;
            Sb[(size_t)m * Mf + ncol] = acc[t][r2];
        }
}

// Row causal softmax -> f16 probabilities (zeros above diagonal).
__global__ void k_softmax(const float* __restrict__ S, _Float16* __restrict__ P, int Mf)
{
    int row = blockIdx.x;                          // b*Mf + m
    int b = row / Mf, m = row % Mf;
    const float* Sr = S + (size_t)b * Mf * Mf + (size_t)m * Mf;
    _Float16*    Pr = P + (size_t)b * Mf * Mf + (size_t)m * Mf;
    int tid = threadIdx.x;
    __shared__ float red[256];
    float mx = -3.0e38f;
    for (int c = tid; c <= m; c += 256) mx = fmaxf(mx, Sr[c]);
    red[tid] = mx; __syncthreads();
    for (int s = 128; s > 0; s >>= 1) { if (tid < s) red[tid] = fmaxf(red[tid], red[tid + s]); __syncthreads(); }
    mx = red[0]; __syncthreads();
    float e[4]; int cnt = 0; float sum = 0.f;
    for (int c = tid; c < Mf; c += 256) {
        float v = (c <= m) ? __expf(Sr[c] - mx) : 0.f;
        e[cnt++] = v; sum += v;
    }
    red[tid] = sum; __syncthreads();
    for (int s = 128; s > 0; s >>= 1) { if (tid < s) red[tid] += red[tid + s]; __syncthreads(); }
    float inv = 1.f / red[0];
    cnt = 0;
    for (int c = tid; c < Mf; c += 256) Pr[c] = (_Float16)(e[cnt++] * inv);
}

// out_folded = P @ Vf + Vf(residual), unfold-scattered into cat; 2 c-tiles/wave.
__global__ void k_pv(const _Float16* __restrict__ P, const _Float16* __restrict__ vfT,
                     const float* __restrict__ kvsrc, float* __restrict__ cat,
                     int fold, int Mf, int Kf)
{
    int wave = blockIdx.x * 8 + (threadIdx.x >> 5);
    int nct2 = Kf >> 5, nmt = Mf >> 4;             // nct/2
    int per_b = nmt * nct2;
    int b = wave / per_b, r = wave % per_b;
    int mt = r / nct2, ct0 = (r % nct2) * 2;
    const _Float16* Pb = P + (size_t)b * Mf * Mf;
    const _Float16* Vb = vfT + (size_t)b * (size_t)Mf * Kf;
    v8f acc[2] = {};
    wmma_tiles<2>(Pb, Mf, mt * 16, Vb, Mf, ct0 * 16, Mf >> 5, acc);
    int lane = threadIdx.x & 31, lo = lane & 15, hi = lane >> 4;
#pragma unroll
    for (int t = 0; t < 2; ++t)
#pragma unroll
        for (int r2 = 0; r2 < 8; ++r2) {
            int m = mt * 16 + r2 + hi * 8, col = (ct0 + t) * 16 + lo;
            int idx = fold_src_index(fold, b, m, col);
            cat[idx] = acc[t][r2] + kvsrc[idx];    // residual +tv
        }
}

// ---------------------------------------------------------------------------
// LayerNorm over 128 channels, 16 tokens per block -> f16 token-major
// ---------------------------------------------------------------------------
__global__ void k_ln(const float* __restrict__ cat, const float* __restrict__ g,
                     const float* __restrict__ bta, _Float16* __restrict__ ln_h)
{
    int blk = blockIdx.x;                          // 100 * 64
    int bn = blk / 64;
    int hw0 = (blk % 64) * 16;
    int b = bn / 25, n = bn % 25;
    size_t base = ((size_t)(b * 128) * 25 + n) * 1024 + hw0;
    __shared__ float tile[16][132];
    int tid = threadIdx.x;
    for (int i = tid; i < 128 * 16; i += 256) {
        int ch = i >> 4, tok = i & 15;
        tile[tok][ch] = cat[base + (size_t)ch * 25600 + tok];
    }
    __syncthreads();
    __shared__ float mu_s[16], rs_s[16];
    if (tid < 16) {
        float s = 0.f;
        for (int ch = 0; ch < 128; ++ch) s += tile[tid][ch];
        float mu = s * (1.f / 128.f);
        float v = 0.f;
        for (int ch = 0; ch < 128; ++ch) { float d = tile[tid][ch] - mu; v += d * d; }
        mu_s[tid] = mu; rs_s[tid] = rsqrtf(v * (1.f / 128.f) + 1e-5f);
    }
    __syncthreads();
    size_t t0 = (size_t)bn * 1024 + hw0;
    for (int i = tid; i < 128 * 16; i += 256) {
        int tok = i >> 7, ch = i & 127;
        float x = (tile[tok][ch] - mu_s[tok]) * rs_s[tok] * g[ch] + bta[ch];
        ln_h[(t0 + tok) * 128 + ch] = (_Float16)x;
    }
}

// hidden = relu(ln @ Wm1); Wm1^T in LDS. Block: 4 mtiles x 2 ngroups. Grid 1600.
__global__ void k_mlp1(const _Float16* __restrict__ ln_h, const _Float16* __restrict__ w1t,
                       _Float16* __restrict__ hidden)
{
    __shared__ __align__(16) _Float16 swb[128 * 128];
    stage_b_lds(swb, w1t, 128 * 128);
    int wv = threadIdx.x >> 5;
    int mt = blockIdx.x * 4 + (wv & 3);
    int ng = wv >> 2;
    v8f acc[4] = {};
    wmma_tiles<4>(ln_h, 128, mt * 16, swb, 128, ng * 64, 4, acc);
    int lane = threadIdx.x & 31, lo = lane & 15, hi = lane >> 4;
#pragma unroll
    for (int t = 0; t < 4; ++t) {
        int ch = ng * 64 + t * 16 + lo;
#pragma unroll
        for (int r = 0; r < 8; ++r) {
            int tk = mt * 16 + r + hi * 8;
            hidden[(size_t)tk * 128 + ch] = (_Float16)fmaxf(acc[t][r], 0.f);
        }
    }
}

// zr = hidden @ Wm2 + ct (ct gathered from cat); Wm2^T in LDS. Grid 1600.
__global__ void k_mlp2(const _Float16* __restrict__ hidden, const _Float16* __restrict__ w2t,
                       const float* __restrict__ cat, _Float16* __restrict__ zr)
{
    __shared__ __align__(16) _Float16 swb[128 * 128];
    stage_b_lds(swb, w2t, 128 * 128);
    int wv = threadIdx.x >> 5;
    int mt = blockIdx.x * 4 + (wv & 3);
    int ng = wv >> 2;
    v8f acc[4] = {};
    wmma_tiles<4>(hidden, 128, mt * 16, swb, 128, ng * 64, 4, acc);
    int lane = threadIdx.x & 31, lo = lane & 15, hi = lane >> 4;
#pragma unroll
    for (int t = 0; t < 4; ++t) {
        int ch = ng * 64 + t * 16 + lo;
#pragma unroll
        for (int r = 0; r < 8; ++r) {
            int tk = mt * 16 + r + hi * 8;
            int b = tk / 25600, n = (tk >> 10) % 25, hw = tk & 1023;
            int cidx = ((b * 128 + ch) * 25 + n) * 1024 + hw;
            zr[(size_t)tk * 128 + ch] = (_Float16)(acc[t][r] + cat[cidx]);
        }
    }
}

// out(b,64,n,h,w) = zr @ W_out; W_out^T in LDS. Block: 8 mtiles x NT=4. Grid 800.
__global__ void k_out(const _Float16* __restrict__ zr, const _Float16* __restrict__ wot,
                      float* __restrict__ out)
{
    __shared__ __align__(16) _Float16 swb[64 * 128];
    stage_b_lds(swb, wot, 64 * 128);
    int wv = threadIdx.x >> 5;
    int mt = blockIdx.x * 8 + wv;
    v8f acc[4] = {};
    wmma_tiles<4>(zr, 128, mt * 16, swb, 128, 0, 4, acc);
    int lane = threadIdx.x & 31, lo = lane & 15, hi = lane >> 4;
#pragma unroll
    for (int t = 0; t < 4; ++t) {
        int ch = t * 16 + lo;
#pragma unroll
        for (int r = 0; r < 8; ++r) {
            int tk = mt * 16 + r + hi * 8;
            int b = tk / 25600, n = (tk >> 10) % 25, hw = tk & 1023;
            out[(size_t)((b * 64 + ch) * 25 + n) * 1024 + hw] = acc[t][r];
        }
    }
}

// ---------------------------------------------------------------------------
extern "C" void kernel_launch(void* const* d_in, const int* in_sizes, int n_in,
                              void* d_out, int out_size, void* d_ws, size_t ws_size,
                              hipStream_t stream)
{
    (void)in_sizes; (void)n_in; (void)out_size;
    const float* x_lf  = (const float*)d_in[0];
    const float* x_hf  = (const float*)d_in[1];
    const float* W_in  = (const float*)d_in[2];
    const float* W_kv  = (const float*)d_in[3];
    const float* ln_g  = (const float*)d_in[4];
    const float* ln_b  = (const float*)d_in[5];
    const float* W_m1  = (const float*)d_in[6];
    const float* W_m2  = (const float*)d_in[7];
    const float* W_out = (const float*)d_in[8];
    float* out = (float*)d_out;
    char* ws = (char*)d_ws;

    size_t off = 0;
    auto take = [&](size_t bytes) { size_t r = off; off += (bytes + 255) & ~(size_t)255; return r; };
    size_t o_q    = take(52428800);                 // q  (b,128,25,1024) f32
    size_t o_kv   = take(52428800);                 // kv f32
    size_t o_cat  = take(52428800);                 // attention output concat f32
    size_t o_xtok = take(26214400);                 // 2x token-major f16 x; later: MLP hidden
    size_t o_wi   = take(16384);
    size_t o_wk   = take(16384);
    size_t o_w1   = take(32768);
    size_t o_w2   = take(32768);
    size_t o_wo   = take(16384);
    size_t o_attn = take(50331648);                 // attention scratch; later ln_h / zr
    if (ws_size < off) return;                      // not enough workspace

    float* q   = (float*)(ws + o_q);
    float* kv  = (float*)(ws + o_kv);
    float* cat = (float*)(ws + o_cat);
    _Float16* xtok_hf = (_Float16*)(ws + o_xtok);
    _Float16* xtok_lf = (_Float16*)(ws + o_xtok + 13107200);
    _Float16* hidden  = (_Float16*)(ws + o_xtok);   // reuse after projections
    _Float16* wt_in = (_Float16*)(ws + o_wi);
    _Float16* wt_kv = (_Float16*)(ws + o_wk);
    _Float16* w1t   = (_Float16*)(ws + o_w1);
    _Float16* w2t   = (_Float16*)(ws + o_w2);
    _Float16* wot   = (_Float16*)(ws + o_wo);
    char* attn = ws + o_attn;
    _Float16* ln_h = (_Float16*)attn;               // after attention done
    _Float16* zr   = (_Float16*)attn;               // overwrites ln_h in mlp2

    // 1) weight transposes (f32 -> f16 N x K row-major)
    k_transpose_f16<<<32, 256, 0, stream>>>(W_in,  wt_in, 64, 128);
    k_transpose_f16<<<32, 256, 0, stream>>>(W_kv,  wt_kv, 64, 128);
    k_transpose_f16<<<64, 256, 0, stream>>>(W_m1,  w1t, 128, 128);
    k_transpose_f16<<<64, 256, 0, stream>>>(W_m2,  w2t, 128, 128);
    k_transpose_f16<<<32, 256, 0, stream>>>(W_out, wot, 128, 64);

    // 2) token gather + projections (weights TDM-staged to LDS)
    k_tok<<<25600, 256, 0, stream>>>(x_hf, xtok_hf);
    k_tok<<<25600, 256, 0, stream>>>(x_lf, xtok_lf);
    k_proj<<<1600, 256, 0, stream>>>(xtok_hf, wt_in, q);
    k_proj<<<1600, 256, 0, stream>>>(xtok_lf, wt_kv, kv);

    // 3) three folded causal attentions
    const int MFs[3] = {1024, 160, 160};
    const int KFs[3] = {800, 5120, 10240};
    for (int f = 0; f < 3; ++f) {
        int Mf = MFs[f], Kf = KFs[f];
        size_t qsz = (size_t)4 * Mf * Kf * 2;
        _Float16* qf  = (_Float16*)attn;
        _Float16* kf  = (_Float16*)(attn + qsz);
        _Float16* vfT = (_Float16*)(attn + 2 * qsz);
        float*    S   = (float*)(attn + 3 * qsz);
        _Float16* P   = (_Float16*)(attn + 3 * qsz + (size_t)4 * Mf * Mf * 4);
        k_fold<<<4 * Mf, 256, 0, stream>>>(q,  qf,  f, Mf, Kf, 1, 0);
        k_fold<<<4 * Mf, 256, 0, stream>>>(kv, kf,  f, Mf, Kf, 1, 0);
        k_fold<<<4 * Mf, 256, 0, stream>>>(kv, vfT, f, Mf, Kf, 0, 1);
        int nmt = Mf / 16, nct = Kf / 16;
        k_scores<<<(4 * nmt * (nmt / 2)) / 8, 256, 0, stream>>>(qf, kf, S, Mf, Kf);
        k_softmax<<<4 * Mf, 256, 0, stream>>>(S, P, Mf);
        k_pv<<<(4 * nmt * (nct / 2)) / 8, 256, 0, stream>>>(P, vfT, kv, cat, f, Mf, Kf);
    }

    // 4) LayerNorm + MLP + output projection
    k_ln<<<6400, 256, 0, stream>>>(cat, ln_g, ln_b, ln_h);
    k_mlp1<<<1600, 256, 0, stream>>>(ln_h, w1t, hidden);
    k_mlp2<<<1600, 256, 0, stream>>>(hidden, w2t, cat, zr);
    k_out<<<800, 256, 0, stream>>>(zr, wot, out);
}
